// GATConvNoSkipsNet_35716948034094
// MI455X (gfx1250) — matrix operand order repaired and checked by reading, not compiled, not run on hardware
//
#include <hip/hip_runtime.h>
#include <hip/hip_bf16.h>

// ---------------------------------------------------------------------------
// GAT (3 layers) for gfx1250.  Edge pipeline: segment-max -> exp/segment-sum
// -> weighted scatter (atomics; node state resident in 192MB L2).
// Layer-2 GEMM (50000x128 @ 128x128) on v_wmma_f32_16x16x32_f16 with
// fragment-packed B (contiguous b128 loads) and vectorized A loads.
// ---------------------------------------------------------------------------

#define N_NODES 50000
#define N_E0    800000
#define N_ETOT  (N_E0 + N_NODES)
#define NEG_SLOPE 0.2f

typedef __attribute__((ext_vector_type(16))) _Float16 v16h;
typedef __attribute__((ext_vector_type(8)))  _Float16 v8h;
typedef __attribute__((ext_vector_type(8)))  float    v8f;

// ---- order-monotone float<->uint encoding for atomicMax on floats ----------
__device__ __forceinline__ unsigned fenc(float f) {
    unsigned b = __float_as_uint(f);
    return (b & 0x80000000u) ? ~b : (b | 0x80000000u);
}
__device__ __forceinline__ float fdec(unsigned e) {
    unsigned b = (e & 0x80000000u) ? (e & 0x7FFFFFFFu) : ~e;
    return __uint_as_float(b);
}
__device__ __forceinline__ void edge_ends(const long long* __restrict__ esrc,
                                          const long long* __restrict__ edst,
                                          int e, long long& s, long long& d) {
    if (e < N_E0) { s = esrc[e]; d = edst[e]; }
    else          { s = d = (long long)(e - N_E0); }   // self loops
}

// ---------------------------- utility fills ---------------------------------
__global__ void k_fill_f32(float* __restrict__ p, float v, int n) {
    int i = blockIdx.x * blockDim.x + threadIdx.x;
    if (i < n) p[i] = v;
}
__global__ void k_fill_u32(unsigned* __restrict__ p, unsigned v, int n) {
    int i = blockIdx.x * blockDim.x + threadIdx.x;
    if (i < n) p[i] = v;
}

// ---------------------------- layer 1: h = x * W1 ---------------------------
__global__ void k_layer1(const float* __restrict__ x, const float* __restrict__ W1,
                         float* __restrict__ h) {
    int i = blockIdx.x * blockDim.x + threadIdx.x;      // over N*128
    if (i >= N_NODES * 128) return;
    int n = i >> 7, c = i & 127;
    h[i] = x[n] * W1[c];
}

// ------------------- attention scores a_src/a_dst [N,H] ---------------------
__global__ void k_attn(const float* __restrict__ h,
                       const float* __restrict__ att_s, const float* __restrict__ att_d,
                       float* __restrict__ a_s, float* __restrict__ a_d,
                       int H, int C) {
    int i = blockIdx.x * blockDim.x + threadIdx.x;      // over N*H
    if (i >= N_NODES * H) return;
    int n = i / H, hh = i - n * H;
    const float* hp = h + (size_t)n * H * C + (size_t)hh * C;
    float as = 0.f, ad = 0.f;
    for (int c = 0; c < C; ++c) {
        float v = hp[c];
        as += v * att_s[hh * C + c];
        ad += v * att_d[hh * C + c];
    }
    a_s[i] = as; a_d[i] = ad;
}

// ----------------- edge pass 1: leaky-relu logit, segment max ---------------
__global__ void k_edge_max(const long long* __restrict__ esrc, const long long* __restrict__ edst,
                           const float* __restrict__ a_s, const float* __restrict__ a_d,
                           unsigned* __restrict__ mEnc, int H) {
    int i = blockIdx.x * blockDim.x + threadIdx.x;      // over Etot*H
    if (i >= N_ETOT * H) return;
    int e = i / H, hh = i - e * H;
    long long s, d; edge_ends(esrc, edst, e, s, d);
    float v = a_s[s * H + hh] + a_d[d * H + hh];
    v = (v > 0.f) ? v : NEG_SLOPE * v;
    atomicMax(&mEnc[d * H + hh], fenc(v));
}

// ----------------- edge pass 2: exp(e - max), segment sum -------------------
__global__ void k_edge_expsum(const long long* __restrict__ esrc, const long long* __restrict__ edst,
                              const float* __restrict__ a_s, const float* __restrict__ a_d,
                              const unsigned* __restrict__ mEnc,
                              float* __restrict__ expE, float* __restrict__ ssum, int H) {
    int i = blockIdx.x * blockDim.x + threadIdx.x;
    if (i >= N_ETOT * H) return;
    int e = i / H, hh = i - e * H;
    long long s, d; edge_ends(esrc, edst, e, s, d);
    float v = a_s[s * H + hh] + a_d[d * H + hh];
    v = (v > 0.f) ? v : NEG_SLOPE * v;
    float ex = __expf(v - fdec(mEnc[d * H + hh]));
    expE[i] = ex;
    atomicAdd(&ssum[d * H + hh], ex);
}

// ----------------- edge pass 3: alpha-weighted scatter of h[src] ------------
__global__ void k_edge_scatter(const long long* __restrict__ esrc, const long long* __restrict__ edst,
                               const float* __restrict__ hsrc, const float* __restrict__ expE,
                               const float* __restrict__ ssum, float* __restrict__ acc,
                               int H, int C) {
    int HC = H * C;
    long long i = (long long)blockIdx.x * blockDim.x + threadIdx.x;  // over Etot*HC
    if (i >= (long long)N_ETOT * HC) return;
    int e  = (int)(i / HC);
    int ch = (int)(i - (long long)e * HC);
    int hh = ch / C;
    long long s, d; edge_ends(esrc, edst, e, s, d);
    float alpha = expE[(long long)e * H + hh] / (ssum[d * H + hh] + 1e-16f);
    atomicAdd(&acc[d * (long long)HC + ch], hsrc[s * (long long)HC + ch] * alpha);
}

// ----------------- bias + relu; optional f32 and f16 outputs ----------------
__global__ void k_bias_relu(const float* __restrict__ acc, const float* __restrict__ bias,
                            float* __restrict__ xf32, _Float16* __restrict__ xf16, int F) {
    int i = blockIdx.x * blockDim.x + threadIdx.x;      // over N*F
    if (i >= N_NODES * F) return;
    int c = i % F;
    float v = acc[i] + bias[c];
    v = v > 0.f ? v : 0.f;
    if (xf32) xf32[i] = v;
    if (xf16) xf16[i] = (_Float16)v;
}

// --- convert W2 (f32 row-major [128,128]) to f16 in WMMA-fragment order -----
// Packed layout: Wp[((nt*4 + ks)*32 + lane)*16 + j]
//   = W[(ks*32 + bkb + j)*128 + nt*16 + (lane&15)],  bkb = (lane<16) ? 0 : 16
// so each lane's whole B fragment is 16 contiguous halfs (32B).
__global__ void k_pack_w(const float* __restrict__ W, _Float16* __restrict__ Wp) {
    int i = blockIdx.x * blockDim.x + threadIdx.x;      // over 8*4*32*16 = 16384
    if (i >= 16384) return;
    int j    = i & 15;
    int lane = (i >> 4) & 31;
    int ks   = (i >> 9) & 3;
    int nt   = (i >> 11) & 7;
    int bkb  = (lane < 16) ? 0 : 16;
    Wp[i] = (_Float16)W[(ks * 32 + bkb + j) * 128 + nt * 16 + (lane & 15)];
}

// ----------------- layer-2 GEMM: Y[N,128] = Xh[N,128] @ Wp ------------------
// 4 waves/block; each wave owns one 16-row tile: 8 N-tiles x 4 K-steps of
// v_wmma_f32_16x16x32_f16.  A: 2x b128 vector loads per fragment.
// B: fragment-packed -> 2x b128 contiguous loads per fragment.
__global__ __launch_bounds__(128) void k_gemm_wmma(const _Float16* __restrict__ X,
                                                   const _Float16* __restrict__ Wp,
                                                   float* __restrict__ Y) {
    const int wave = threadIdx.x >> 5;
    const int lane = threadIdx.x & 31;
    const int row0 = (blockIdx.x * 4 + wave) * 16;
    const int arow = row0 + (lane & 15);
    const int akb  = (lane < 16) ? 0 : 16;   // A K-base (contiguous K, matches packed B)

    // preload all 4 A fragments (K = 0..127) for this 16-row tile
    v16h afrag[4];
#pragma unroll
    for (int ks = 0; ks < 4; ++ks) {
        const _Float16* ap = X + (size_t)arow * 128 + ks * 32 + akb;
        v8h lo = *(const v8h*)(ap);        // K = kb+0..7   (16B aligned)
        v8h hi = *(const v8h*)(ap + 8);    // K = kb+8..15  (16B aligned)
        afrag[ks] = __builtin_shufflevector(lo, hi, 0, 1, 2, 3, 4, 5, 6, 7,
                                            8, 9, 10, 11, 12, 13, 14, 15);
    }

#pragma unroll
    for (int nt = 0; nt < 8; ++nt) {
        v8f c = {};
#pragma unroll
        for (int ks = 0; ks < 4; ++ks) {
            v16h b = *(const v16h*)(Wp + (size_t)((nt * 4 + ks) * 32 + lane) * 16);
            c = __builtin_amdgcn_wmma_f32_16x16x32_f16(
                    false, afrag[ks], false, b, (short)0, c, false, false);
        }
        const int mbase = (lane < 16) ? 0 : 8;   // C/D layout: VGPR i -> M = mbase+i
#pragma unroll
        for (int i = 0; i < 8; ++i)
            Y[(size_t)(row0 + mbase + i) * 128 + nt * 16 + (lane & 15)] = c[i];
    }
}

// ----------------- layer 3: h3 = X3 @ W3 (128->1); scalar attn --------------
__global__ void k_layer3_h(const float* __restrict__ X3, const float* __restrict__ W3,
                           const float* __restrict__ att_s, const float* __restrict__ att_d,
                           float* __restrict__ h3, float* __restrict__ a_s, float* __restrict__ a_d) {
    int n = blockIdx.x * blockDim.x + threadIdx.x;
    if (n >= N_NODES) return;
    const float* xp = X3 + (size_t)n * 128;
    float acc = 0.f;
#pragma unroll 8
    for (int c = 0; c < 128; ++c) acc += xp[c] * W3[c];
    h3[n]  = acc;
    a_s[n] = acc * att_s[0];
    a_d[n] = acc * att_d[0];
}

__global__ void k_final(const float* __restrict__ acc, const float* __restrict__ b3,
                        float* __restrict__ out) {
    int n = blockIdx.x * blockDim.x + threadIdx.x;
    if (n >= N_NODES) return;
    float v = acc[n] + b3[0];
    out[n] = 1.f / (1.f + __expf(-v));
}

// ---------------------------------------------------------------------------
extern "C" void kernel_launch(void* const* d_in, const int* in_sizes, int n_in,
                              void* d_out, int out_size, void* d_ws, size_t ws_size,
                              hipStream_t stream) {
    (void)in_sizes; (void)n_in; (void)out_size; (void)ws_size;

    const float*     x     = (const float*)d_in[0];
    const long long* eidx  = (const long long*)d_in[1];   // int64 [2, E]
    const long long* esrc  = eidx;
    const long long* edst  = eidx + N_E0;
    const float* W1   = (const float*)d_in[3];
    const float* as1  = (const float*)d_in[4];
    const float* ad1  = (const float*)d_in[5];
    const float* b1   = (const float*)d_in[6];
    const float* W2   = (const float*)d_in[7];
    const float* as2  = (const float*)d_in[8];
    const float* ad2  = (const float*)d_in[9];
    const float* b2   = (const float*)d_in[10];
    const float* W3   = (const float*)d_in[11];
    const float* as3  = (const float*)d_in[12];
    const float* ad3  = (const float*)d_in[13];
    const float* b3   = (const float*)d_in[14];

    // workspace carve-up (256B aligned)
    size_t off = 0;
    auto carve = [&](size_t bytes) {
        void* p = (char*)d_ws + off;
        off += (bytes + 255) & ~(size_t)255;
        return p;
    };
    float*     hbuf = (float*)carve((size_t)N_NODES * 128 * 4); // h / X (ping)
    float*     accb = (float*)carve((size_t)N_NODES * 128 * 4); // aggregation acc
    _Float16*  Xh   = (_Float16*)carve((size_t)N_NODES * 128 * 2);
    _Float16*  Wp   = (_Float16*)carve((size_t)128 * 128 * 2);  // fragment-packed
    float*     a_s  = (float*)carve((size_t)N_NODES * 2 * 4);
    float*     a_d  = (float*)carve((size_t)N_NODES * 2 * 4);
    unsigned*  mE   = (unsigned*)carve((size_t)N_NODES * 2 * 4);
    float*     ss   = (float*)carve((size_t)N_NODES * 2 * 4);
    float*     expE = (float*)carve((size_t)N_ETOT * 2 * 4);
    float*     h3   = (float*)carve((size_t)N_NODES * 4);

    const int B = 256;
    auto g = [](long long n, int b) { return (unsigned)((n + b - 1) / b); };
    const unsigned ENC_NEG_INF = 0x007FFFFFu;  // fenc(-inf)

    // =================== layer 1 (1 -> 2x64, concat) =======================
    k_layer1<<<g((long long)N_NODES * 128, B), B, 0, stream>>>(x, W1, hbuf);
    k_attn<<<g((long long)N_NODES * 2, B), B, 0, stream>>>(hbuf, as1, ad1, a_s, a_d, 2, 64);
    k_fill_u32<<<g(N_NODES * 2, B), B, 0, stream>>>(mE, ENC_NEG_INF, N_NODES * 2);
    k_fill_f32<<<g(N_NODES * 2, B), B, 0, stream>>>(ss, 0.f, N_NODES * 2);
    k_fill_f32<<<g((long long)N_NODES * 128, B), B, 0, stream>>>(accb, 0.f, N_NODES * 128);
    k_edge_max<<<g((long long)N_ETOT * 2, B), B, 0, stream>>>(esrc, edst, a_s, a_d, mE, 2);
    k_edge_expsum<<<g((long long)N_ETOT * 2, B), B, 0, stream>>>(esrc, edst, a_s, a_d, mE, expE, ss, 2);
    k_edge_scatter<<<g((long long)N_ETOT * 128, B), B, 0, stream>>>(esrc, edst, hbuf, expE, ss, accb, 2, 64);
    // X2 only needed as f16 (feeds WMMA GEMM)
    k_bias_relu<<<g((long long)N_NODES * 128, B), B, 0, stream>>>(accb, b1, nullptr, Xh, 128);

    // =================== layer 2 (128 -> 2x64, concat) =====================
    k_pack_w<<<g(16384, B), B, 0, stream>>>(W2, Wp);
    k_gemm_wmma<<<N_NODES / 64, 128, 0, stream>>>(Xh, Wp, hbuf);   // h2 -> hbuf
    k_attn<<<g((long long)N_NODES * 2, B), B, 0, stream>>>(hbuf, as2, ad2, a_s, a_d, 2, 64);
    k_fill_u32<<<g(N_NODES * 2, B), B, 0, stream>>>(mE, ENC_NEG_INF, N_NODES * 2);
    k_fill_f32<<<g(N_NODES * 2, B), B, 0, stream>>>(ss, 0.f, N_NODES * 2);
    k_fill_f32<<<g((long long)N_NODES * 128, B), B, 0, stream>>>(accb, 0.f, N_NODES * 128);
    k_edge_max<<<g((long long)N_ETOT * 2, B), B, 0, stream>>>(esrc, edst, a_s, a_d, mE, 2);
    k_edge_expsum<<<g((long long)N_ETOT * 2, B), B, 0, stream>>>(esrc, edst, a_s, a_d, mE, expE, ss, 2);
    k_edge_scatter<<<g((long long)N_ETOT * 128, B), B, 0, stream>>>(esrc, edst, hbuf, expE, ss, accb, 2, 64);
    // X3 needed in f32 for layer-3 dot products
    k_bias_relu<<<g((long long)N_NODES * 128, B), B, 0, stream>>>(accb, b2, hbuf, nullptr, 128);

    // =================== layer 3 (128 -> 1, 1 head) ========================
    k_layer3_h<<<g(N_NODES, B), B, 0, stream>>>(hbuf, W3, as3, ad3, h3, a_s, a_d);
    k_fill_u32<<<g(N_NODES, B), B, 0, stream>>>(mE, ENC_NEG_INF, N_NODES);
    k_fill_f32<<<g(N_NODES, B), B, 0, stream>>>(ss, 0.f, N_NODES);
    k_fill_f32<<<g(N_NODES, B), B, 0, stream>>>(accb, 0.f, N_NODES);
    k_edge_max<<<g((long long)N_ETOT, B), B, 0, stream>>>(esrc, edst, a_s, a_d, mE, 1);
    k_edge_expsum<<<g((long long)N_ETOT, B), B, 0, stream>>>(esrc, edst, a_s, a_d, mE, expE, ss, 1);
    k_edge_scatter<<<g((long long)N_ETOT, B), B, 0, stream>>>(esrc, edst, h3, expE, ss, accb, 1, 1);
    k_final<<<g(N_NODES, B), B, 0, stream>>>(accb, b3, (float*)d_out);
}
